// MultiHeadAttention_68152541053100
// MI455X (gfx1250) — matrix operand order repaired and checked by reading, not compiled, not run on hardware
//
#include <hip/hip_runtime.h>

// ---------------- CDNA5 (gfx1250) WMMA multi-head attention ----------------
// B=4, S=2048, D=1024, H=16, PD=64.  wave32, WMMA f32_16x16x32_bf16.

#define Bc 4
#define Sc 2048
#define Dc 1024
#define Hc 16
#define PDc 64
#define Mc (Bc * Sc)   // 8192

typedef __attribute__((ext_vector_type(16))) __bf16 v16bf;
typedef __attribute__((ext_vector_type(8)))  float  v8f;
typedef int v4i_ __attribute__((vector_size(16)));

union FragAB { v16bf v; unsigned u[8]; };

#if defined(__gfx1250__) && __has_builtin(__builtin_amdgcn_global_load_async_to_lds_b128)
#define ATTN_ASYNC 1
#else
#define ATTN_ASYNC 0
#endif

__device__ __forceinline__ unsigned short f2bf(float f) {
  unsigned x = __float_as_uint(f);
  unsigned r = x + 0x7fffu + ((x >> 16) & 1u);   // round-to-nearest-even
  return (unsigned short)(r >> 16);
}

__device__ __forceinline__ v8f wmma_bf16(const FragAB& a, const FragAB& b, v8f c) {
  return __builtin_amdgcn_wmma_f32_16x16x32_bf16(false, a.v, false, b.v,
                                                 (short)0, c, false, false);
}

// Copy 32 contiguous bytes global -> LDS.  Async (GLOBAL_LOAD_ASYNC_TO_LDS_B128,
// ASYNCcnt) when available; manual b64 copies otherwise.  Caller guarantees
// 16-byte alignment of both addresses.
__device__ __forceinline__ void stage32B(unsigned short* ldst,
                                         const unsigned short* gsrc) {
#if ATTN_ASYNC
  v4i_ __attribute__((address_space(3)))* l =
      (v4i_ __attribute__((address_space(3)))*)(unsigned)(uintptr_t)ldst;
  v4i_ __attribute__((address_space(1)))* g =
      (v4i_ __attribute__((address_space(1)))*)(uintptr_t)gsrc;
  __builtin_amdgcn_global_load_async_to_lds_b128(g, l, 0, 0);
  __builtin_amdgcn_global_load_async_to_lds_b128(g, l, 16, 0);
#else
  const uint2* s = (const uint2*)gsrc;
  uint2* d = (uint2*)ldst;
  d[0] = s[0]; d[1] = s[1]; d[2] = s[2]; d[3] = s[3];
#endif
}

__device__ __forceinline__ void wait_async0() {
#if ATTN_ASYNC
#if __has_builtin(__builtin_amdgcn_s_wait_asynccnt)
  __builtin_amdgcn_s_wait_asynccnt(0);
#else
  asm volatile("s_wait_asynccnt 0x0" ::: "memory");
#endif
#endif
}

// ---------------------------------------------------------------------------
// GEMM: Out[M,N] = A[M,K] * W[K,N] + bias.  K = N = 1024, M = 8192.
// A_IS_BF16: A is bf16 bits (ushort), else fp32 (converted on LDS stage).
// OUT_MODE: 0 = fp32 row-major [M,N]; 1 = bf16 [B,H,S,PD]; 2 = bf16 [B,H,PD,S].
// ---------------------------------------------------------------------------
template <bool A_IS_BF16, int OUT_MODE>
__global__ __launch_bounds__(128) void gemm_wmma_bf16(
    const void* __restrict__ Aptr, const float* __restrict__ W,
    const float* __restrict__ bias, void* __restrict__ Out) {
  __shared__ unsigned short Alds[64][36];  // 64 rows x 32 K, padded
  __shared__ unsigned short Wt[64][36];    // 64 cols(N) x 32 K, transposed

  const int tid  = threadIdx.x;
  const int wave = tid >> 5, lane = tid & 31;
  const int tm = blockIdx.x, tn = blockIdx.y;

  v8f acc[4] = {};
  const int arow  = (wave << 4) | (lane & 15);
  const int khalf = (lane >> 4) << 3;   // 0 or 8 (unused)
  const int kb    = (lane >> 4) << 4;   // B: 0 or 16

  for (int k0 = 0; k0 < Dc; k0 += 32) {
    // ---- stage A tile 64x32 into LDS (bf16) ----
    {
      const int r  = tid >> 1;
      const int cb = (tid & 1) << 4;  // 0 or 16
      if (A_IS_BF16) {
        const unsigned short* A = (const unsigned short*)Aptr;
        const uint2* src = (const uint2*)(A + (size_t)(tm * 64 + r) * Dc + k0 + cb);
        uint2* dst = (uint2*)&Alds[r][cb];
#pragma unroll
        for (int i = 0; i < 4; ++i) dst[i] = src[i];
      } else {
        const float* A = (const float*)Aptr;
        const float* src = A + (size_t)(tm * 64 + r) * Dc + k0 + cb;
        if (k0 + 32 < Dc) __builtin_prefetch(src + 32, 0, 1);
#pragma unroll
        for (int i = 0; i < 4; ++i) {
          float4 f = ((const float4*)src)[i];
          unsigned p0 = (unsigned)f2bf(f.x) | ((unsigned)f2bf(f.y) << 16);
          unsigned p1 = (unsigned)f2bf(f.z) | ((unsigned)f2bf(f.w) << 16);
          unsigned* d = (unsigned*)&Alds[r][cb + i * 4];
          d[0] = p0; d[1] = p1;
        }
      }
    }
    // ---- stage W tile 32x64, transposed into LDS ----
    {
      const int kr = tid >> 2;
      const int nb = (tid & 3) << 4;
      const float* src = W + (size_t)(k0 + kr) * Dc + tn * 64 + nb;
      if (k0 + 32 < Dc) __builtin_prefetch(src + 32 * Dc, 0, 1);
#pragma unroll
      for (int i = 0; i < 4; ++i) {
        float4 f = ((const float4*)src)[i];
        Wt[nb + i * 4 + 0][kr] = f2bf(f.x);
        Wt[nb + i * 4 + 1][kr] = f2bf(f.y);
        Wt[nb + i * 4 + 2][kr] = f2bf(f.z);
        Wt[nb + i * 4 + 3][kr] = f2bf(f.w);
      }
    }
    __syncthreads();

    // ---- A fragment (16x32, k = 16*(lane/16) + s) ----
    FragAB a;
#pragma unroll
    for (int p = 0; p < 4; ++p)
      a.u[p] = *(const unsigned*)&Alds[arow][kb + 2 * p];
#pragma unroll
    for (int p = 0; p < 4; ++p)
      a.u[4 + p] = *(const unsigned*)&Alds[arow][kb + 8 + 2 * p];

    // ---- 4 column tiles: B fragment (32x16) + WMMA ----
#pragma unroll
    for (int c = 0; c < 4; ++c) {
      FragAB bfr;
      const int n = (c << 4) | (lane & 15);
#pragma unroll
      for (int p = 0; p < 8; ++p)
        bfr.u[p] = *(const unsigned*)&Wt[n][kb + 2 * p];
      acc[c] = wmma_bf16(a, bfr, acc[c]);
    }
    __syncthreads();
  }

  // ---- store C (lane = col, VGPR r = row r + 8*(lane>=16)) ----
#pragma unroll
  for (int c = 0; c < 4; ++c) {
    const int ncol = tn * 64 + (c << 4) + (lane & 15);
    const float bv = bias[ncol];
#pragma unroll
    for (int r = 0; r < 8; ++r) {
      const int mrow = tm * 64 + (wave << 4) + r + ((lane >> 4) << 3);
      const float val = acc[c][r] + bv;
      if (OUT_MODE == 0) {
        ((float*)Out)[(size_t)mrow * Dc + ncol] = val;
      } else {
        unsigned short* O = (unsigned short*)Out;
        const int bb = mrow >> 11, s = mrow & (Sc - 1);
        const int hh = ncol >> 6, pd = ncol & (PDc - 1);
        if (OUT_MODE == 1)
          O[(((size_t)(bb * Hc + hh) * Sc) + s) * PDc + pd] = f2bf(val);
        else  // per-head transposed: [B,H,PD,S]
          O[(((size_t)(bb * Hc + hh) * PDc) + pd) * Sc + s] = f2bf(val);
      }
    }
  }
}

// ---------------------------------------------------------------------------
// Flash-style attention per (b,h): 64 query rows per workgroup (4 waves x 16),
// streaming 32-key tiles with double-buffered (async) K/V staging.
// V comes in pre-transposed per head ([B,H,PD,S]) so its tile stages as
// contiguous 32B rows.  Reference "mask" only adds log(1e-6+1) ~ 1e-6 on the
// lower triangle, so all key tiles are visited regardless.
// ---------------------------------------------------------------------------
__global__ __launch_bounds__(128) void attn_wmma_bf16(
    const unsigned short* __restrict__ Qh, const unsigned short* __restrict__ Kh,
    const unsigned short* __restrict__ VTh, unsigned short* __restrict__ Oc,
    const int* __restrict__ causal_flag) {
  __shared__ unsigned short Qlds[64][68];        // 64 q-rows x 64 pd
  __shared__ unsigned short Klds[2][32][72];     // 32 keys x 64 pd, dbl-buffered
  __shared__ unsigned short Vt[2][64][40];       // 64 pd x 32 keys, dbl-buffered
  __shared__ unsigned short Plds[4][16][36];     // per-wave P transpose buffer

  const int tid  = threadIdx.x;
  const int wave = tid >> 5, lane = tid & 31;
  const int bh = blockIdx.x;                     // b*16 + h
  const int q0 = blockIdx.y * 64;
  const size_t base = (size_t)bh * Sc * PDc;
  const unsigned short* Qb  = Qh  + base;
  const unsigned short* Kb  = Kh  + base;
  const unsigned short* VTb = VTh + base;        // row stride Sc per pd
  const int causal = *causal_flag;

  // staging coordinates (each thread moves 32B per tile array)
  const int krow = tid >> 2, kcb = (tid & 3) << 4;   // K: 4 thr/row of 64
  const int vrow = tid >> 1, vcb = (tid & 1) << 4;   // Vt: 2 thr/row of 32

  // ---- stage Q tile 64x64 ----
  {
    const int r  = tid >> 1;
    const int cb = (tid & 1) << 5;  // 0 or 32
    const uint2* src = (const uint2*)(Qb + (size_t)(q0 + r) * PDc + cb);
    uint2* dst = (uint2*)&Qlds[r][cb];
#pragma unroll
    for (int i = 0; i < 8; ++i) dst[i] = src[i];
  }
  // ---- stage first K/V tile into buffer 0 ----
  stage32B(&Klds[0][krow][kcb], Kb + (size_t)krow * PDc + kcb);
  stage32B(&Vt[0][vrow][vcb], VTb + (size_t)vrow * Sc + vcb);
  wait_async0();
  __syncthreads();

  const int khalf = (lane >> 4) << 3;
  const int kb    = (lane >> 4) << 4;

  // ---- Q fragments: 16x64 = two 16x32 A-operands ----
  FragAB aq[2];
#pragma unroll
  for (int cc = 0; cc < 2; ++cc) {
    const int row = (wave << 4) + (lane & 15);
#pragma unroll
    for (int p = 0; p < 4; ++p)
      aq[cc].u[p] = *(const unsigned*)&Qlds[row][cc * 32 + kb + 2 * p];
#pragma unroll
    for (int p = 0; p < 4; ++p)
      aq[cc].u[4 + p] = *(const unsigned*)&Qlds[row][cc * 32 + kb + 8 + 2 * p];
  }

  float mrow[8], lrow[8];
  v8f o[4] = {};
#pragma unroll
  for (int r = 0; r < 8; ++r) { mrow[r] = -1e30f; lrow[r] = 0.0f; }

  for (int it = 0; it < Sc / 32; ++it) {
    const int t0 = it * 32;
    const int cur = it & 1;
    // ---- prefetch next tile into the other buffer (overlaps compute) ----
    if (t0 + 32 < Sc) {
      stage32B(&Klds[cur ^ 1][krow][kcb],
               Kb + (size_t)(t0 + 32 + krow) * PDc + kcb);
      stage32B(&Vt[cur ^ 1][vrow][vcb],
               VTb + (size_t)vrow * Sc + (t0 + 32) + vcb);
    }

    // ---- S = Q * K^T  (two 16-key halves, K-dim 64 = 2 chunks) ----
    v8f s[2];
#pragma unroll
    for (int half = 0; half < 2; ++half) {
      const int key = (half << 4) | (lane & 15);
      FragAB bk;
      v8f z = {};
#pragma unroll
      for (int p = 0; p < 8; ++p)
        bk.u[p] = *(const unsigned*)&Klds[cur][key][kb + 2 * p];      // pd 0..31
      z = wmma_bf16(aq[0], bk, z);
#pragma unroll
      for (int p = 0; p < 8; ++p)
        bk.u[p] = *(const unsigned*)&Klds[cur][key][32 + kb + 2 * p]; // pd 32..63
      z = wmma_bf16(aq[1], bk, z);
      s[half] = z;
    }

    // ---- online softmax (row stats across the 16-lane half-wave) ----
#pragma unroll
    for (int r = 0; r < 8; ++r) {
      float v0 = s[0][r] * 0.125f;   // 1/sqrt(64)
      float v1 = s[1][r] * 0.125f;
      if (causal) {
        const int qi  = q0 + (wave << 4) + r + ((lane >> 4) << 3);
        const int ki0 = t0 + (lane & 15);
        v0 += (qi >= ki0)      ? 9.9999950e-7f : 0.0f;  // log(1e-6 + 1)
        v1 += (qi >= ki0 + 16) ? 9.9999950e-7f : 0.0f;
      }
      float tmax = fmaxf(v0, v1);
#pragma unroll
      for (int off = 1; off < 16; off <<= 1)
        tmax = fmaxf(tmax, __shfl_xor(tmax, off, 32));
      const float mnew = fmaxf(mrow[r], tmax);
      const float corr = __expf(mrow[r] - mnew);
      const float p0 = __expf(v0 - mnew);
      const float p1 = __expf(v1 - mnew);
      float rs = p0 + p1;
#pragma unroll
      for (int off = 1; off < 16; off <<= 1) rs += __shfl_xor(rs, off, 32);
      lrow[r] = lrow[r] * corr + rs;
      mrow[r] = mnew;
#pragma unroll
      for (int c = 0; c < 4; ++c) o[c][r] *= corr;
      const int prow = r + ((lane >> 4) << 3);
      Plds[wave][prow][(lane & 15)]      = f2bf(p0);
      Plds[wave][prow][16 + (lane & 15)] = f2bf(p1);
    }
    asm volatile("" ::: "memory");  // same-wave LDS RAW: DS ops are in-order

    // ---- O += P(16x32) * V(32x64) ----
    FragAB ap;
#pragma unroll
    for (int p = 0; p < 4; ++p)
      ap.u[p] = *(const unsigned*)&Plds[wave][lane & 15][kb + 2 * p];
#pragma unroll
    for (int p = 0; p < 4; ++p)
      ap.u[4 + p] = *(const unsigned*)&Plds[wave][lane & 15][kb + 8 + 2 * p];
#pragma unroll
    for (int c = 0; c < 4; ++c) {
      const int pd = (c << 4) | (lane & 15);
      FragAB bv;
#pragma unroll
      for (int p = 0; p < 8; ++p)
        bv.u[p] = *(const unsigned*)&Vt[cur][pd][kb + 2 * p];
      o[c] = wmma_bf16(ap, bv, o[c]);
    }

    wait_async0();     // next tile's DMA complete (this wave)
    __syncthreads();   // all waves done with cur buffer + see new data
  }

  // ---- normalize and write concat layout [B,S,D] as bf16 ----
  const int b = bh >> 4, h = bh & 15;
#pragma unroll
  for (int c = 0; c < 4; ++c) {
    const int pd = (c << 4) + (lane & 15);
#pragma unroll
    for (int r = 0; r < 8; ++r) {
      const int srow = q0 + (wave << 4) + r + ((lane >> 4) << 3);
      const float val = o[c][r] / lrow[r];
      Oc[((size_t)(b * Sc + srow)) * Dc + h * PDc + pd] = f2bf(val);
    }
  }
}

// ---------------------------------------------------------------------------
extern "C" void kernel_launch(void* const* d_in, const int* in_sizes, int n_in,
                              void* d_out, int out_size, void* d_ws,
                              size_t ws_size, hipStream_t stream) {
  const float* q  = (const float*)d_in[0];
  const float* k  = (const float*)d_in[1];
  const float* v  = (const float*)d_in[2];
  const float* Wq = (const float*)d_in[3];
  const float* bq = (const float*)d_in[4];
  const float* Wk = (const float*)d_in[5];
  const float* bk = (const float*)d_in[6];
  const float* Wv = (const float*)d_in[7];
  const float* bv = (const float*)d_in[8];
  const float* Wo = (const float*)d_in[9];
  const float* bo = (const float*)d_in[10];
  const int* causal = (const int*)d_in[11];

  // ws layout: 4 bf16 tensors of M*D = 8192*1024 halves (16 MB each)
  unsigned short* Qh  = (unsigned short*)d_ws;
  unsigned short* Kh  = Qh + (size_t)Mc * Dc;
  unsigned short* VTh = Kh + (size_t)Mc * Dc;   // stored [B,H,PD,S]
  unsigned short* Oc  = VTh + (size_t)Mc * Dc;

  const dim3 gblk(128);
  const dim3 ggrid(Mc / 64, Dc / 64);  // 128 x 16

  gemm_wmma_bf16<false, 1><<<ggrid, gblk, 0, stream>>>(q, Wq, bq, Qh);
  gemm_wmma_bf16<false, 1><<<ggrid, gblk, 0, stream>>>(k, Wk, bk, Kh);
  gemm_wmma_bf16<false, 2><<<ggrid, gblk, 0, stream>>>(v, Wv, bv, VTh);

  attn_wmma_bf16<<<dim3(Bc * Hc, Sc / 64), gblk, 0, stream>>>(Qh, Kh, VTh, Oc,
                                                              causal);

  gemm_wmma_bf16<true, 0><<<ggrid, gblk, 0, stream>>>(Oc, Wo, bo,
                                                      (float*)d_out);
}
